// VanillaAttention_66288525246587
// MI455X (gfx1250) — compile-verified, matrix operand order checked by reading, not run on hardware
//
#include <hip/hip_runtime.h>
#include <hip/hip_bf16.h>

typedef __attribute__((ext_vector_type(16))) _Float16 v16h;
typedef __attribute__((ext_vector_type(8)))  float    v8f;

// Problem constants (from reference)
#define Bc  2
#define Lc  2048
#define Dc  1024
#define NHc 16
#define HDc 64

// Packed region size (in v16h units) for each operand buffer: 8 MB each.
#define PKREG 262144

__device__ __forceinline__ v8f wmma_f16(v16h a, v16h b, v8f c) {
  // D = A*B + C, 16x16x32, f16 in / f32 acc
  return __builtin_amdgcn_wmma_f32_16x16x32_f16(false, a, false, b, (short)0, c,
                                                false, false);
}

// A/B operand element -> K index mapping (16-bit, 16x32 tile, wave32):
// K = (e/8)*16 + (e%8) + (lane/16)*8
__device__ __forceinline__ int kOf(int e, int half) {
  return ((e >> 3) << 4) + (e & 7) + (half << 3);
}

// Inverse: K index w in [0,32) -> (lane-half, element)
__device__ __forceinline__ void kdec(int w, int& half, int& e) {
  half = (w >> 3) & 1;
  e    = ((w >> 4) << 3) + (w & 7);
}

// ---------------------------------------------------------------------------
// Pack x (f32 [4096,1024]) into f16 A-operand tiles:
// layout [rt(256)][kt(32)][lane(32)] : v16h  (32 B contiguous per lane)
// ---------------------------------------------------------------------------
__global__ __launch_bounds__(32) void pack_x_kernel(
    const float* __restrict__ x, v16h* __restrict__ xpk)
{
  const int lane = threadIdx.x, half = lane >> 4, ln = lane & 15;
  const int rt = blockIdx.x, kt = blockIdx.y;
  v16h v;
#pragma unroll
  for (int e = 0; e < 16; ++e)
    v[e] = (_Float16)x[(size_t)(rt * 16 + ln) * Dc + kt * 32 + kOf(e, half)];
  xpk[((size_t)rt * 32 + kt) * 32 + lane] = v;
}

// ---------------------------------------------------------------------------
// Pack the 4 weight matrices (f32 [1024,1024]) into f16 B-operand tiles:
// layout [z(4)][nt(64)][kt(32)][lane(32)] : v16h
// ---------------------------------------------------------------------------
__global__ __launch_bounds__(32) void pack_w_kernel(
    const float* __restrict__ w0, const float* __restrict__ w1,
    const float* __restrict__ w2, const float* __restrict__ w3,
    v16h* __restrict__ wpk)
{
  const int lane = threadIdx.x, half = lane >> 4, ln = lane & 15;
  const int nt = blockIdx.x, kt = blockIdx.y, z = blockIdx.z;
  const float* w = (z == 0) ? w0 : (z == 1) ? w1 : (z == 2) ? w2 : w3;
  v16h v;
#pragma unroll
  for (int e = 0; e < 16; ++e)
    v[e] = (_Float16)w[(size_t)(kt * 32 + kOf(e, half)) * Dc + nt * 16 + ln];
  wpk[(((size_t)z * 64 + nt) * 32 + kt) * 32 + lane] = v;
}

// ---------------------------------------------------------------------------
// Kernel 1a: Q/K projection + bias + RoPE (branch-free hot loop).
// grid.x = 64 (4 row-tiles per wave), grid.y = 64 jobs:
//   [0,32) Q pair-tiles, [32,64) K pair-tiles.
// Pair-tile p: head h = p/2, j = p%2 -> cols [h*64+j*16,+16) and +32 (RoPE
// rotation partners). Output pre-swizzled over HD: [bh][seq16][hi(2)][lane][e]
// ---------------------------------------------------------------------------
__global__ __launch_bounds__(32) void qk_rope_kernel(
    const v16h* __restrict__ xpk, const v16h* __restrict__ wpk,
    const float* __restrict__ bq, const float* __restrict__ bk,
    _Float16* __restrict__ qpk, _Float16* __restrict__ kpk)
{
  const int lane = threadIdx.x, half = lane >> 4, ln = lane & 15;
  const int rt0 = blockIdx.x * 4;
  const int job = blockIdx.y;

  const int mat = job >> 5;          // 0 = Q, 1 = K
  const int p = job & 31;
  const int h = p >> 1, j = p & 1;
  const int colA = h * HDc + j * 16;
  const int colB = colA + 32;
  const float* bias = mat ? bk : bq;
  _Float16* dst = mat ? kpk : qpk;

  const v16h* bbaseA = wpk + ((size_t)mat * 64 + (colA >> 4)) * 32 * 32;
  const v16h* bbaseB = wpk + ((size_t)mat * 64 + (colB >> 4)) * 32 * 32;

  v8f zv = {};
  v8f c0[4] = {zv, zv, zv, zv};
  v8f c1[4] = {zv, zv, zv, zv};

  for (int kt = 0; kt < 32; ++kt) {
    __builtin_prefetch((const void*)&bbaseA[(kt + 1) * 32 + lane], 0, 1);
    __builtin_prefetch((const void*)&bbaseB[(kt + 1) * 32 + lane], 0, 1);
    const v16h b0 = bbaseA[kt * 32 + lane];
    const v16h b1 = bbaseB[kt * 32 + lane];
#pragma unroll
    for (int t = 0; t < 4; ++t) {
      const v16h a = xpk[((size_t)(rt0 + t) * 32 + kt) * 32 + lane];
      c0[t] = wmma_f16(a, b0, c0[t]);
      c1[t] = wmma_f16(a, b1, c1[t]);
    }
  }

  const float bA = bias[colA + ln];
  const float bB = bias[colB + ln];
  const int d = (colA & 63) + ln;                 // rope freq index, 0..31
  const float invts = __powf(10000.0f, -(float)d / 32.0f);
  int hd_, ed_;
  kdec(d, hd_, ed_);                              // same decode for hi=0/1
#pragma unroll
  for (int t = 0; t < 4; ++t) {
#pragma unroll
    for (int i = 0; i < 8; ++i) {
      const int row = (rt0 + t) * 16 + i + half * 8;  // C/D: M = v + 8*(l/16)
      const int b   = row >> 11;
      const int pos = row & (Lc - 1);
      float sn, cs;
      __sincosf((float)pos * invts, &sn, &cs);
      const float f  = c0[t][i] + bA;
      const float s2 = c1[t][i] + bB;
      const float o1 = f * cs - s2 * sn;
      const float o2 = s2 * cs + f * sn;
      const int bh = b * NHc + h;
      // [bh][seq16(128)][hi(2)][lane(32)][e(16)] ; lane = pos%16 + 16*half
      const size_t idx =
          ((((size_t)bh * 128 + (pos >> 4)) * 2) * 32 + (pos & 15) + 16 * hd_) * 16 + ed_;
      dst[idx]       = (_Float16)o1;   // hi = 0 (hd = d)
      dst[idx + 512] = (_Float16)o2;   // hi = 1 (hd = d + 32)
    }
  }
}

// ---------------------------------------------------------------------------
// Kernel 1b: V projection + bias (branch-free hot loop).
// grid.x = 64 (4 row-tiles per wave), grid.y = 64 column tiles.
// Output pre-swizzled as B operand over keys: [bh][kt32][nt(4)][lane][e]
// ---------------------------------------------------------------------------
__global__ __launch_bounds__(32) void v_proj_kernel(
    const v16h* __restrict__ xpk, const v16h* __restrict__ wpk,
    const float* __restrict__ bv, _Float16* __restrict__ vpk)
{
  const int lane = threadIdx.x, half = lane >> 4, ln = lane & 15;
  const int rt0 = blockIdx.x * 4;
  const int ct  = blockIdx.y;
  const int colA = ct * 16;

  const v16h* bbase = wpk + ((size_t)2 * 64 + ct) * 32 * 32;

  v8f zv = {};
  v8f c0[4] = {zv, zv, zv, zv};

  for (int kt = 0; kt < 32; ++kt) {
    __builtin_prefetch((const void*)&bbase[(kt + 1) * 32 + lane], 0, 1);
    const v16h b0 = bbase[kt * 32 + lane];
#pragma unroll
    for (int t = 0; t < 4; ++t) {
      const v16h a = xpk[((size_t)(rt0 + t) * 32 + kt) * 32 + lane];
      c0[t] = wmma_f16(a, b0, c0[t]);
    }
  }

  const float bA = bv[colA + ln];
  const int h  = colA >> 6;
  const int wc = (colA & 63) + ln;   // within-head column, 0..63
  const int ntv = wc >> 4, nlv = wc & 15;
#pragma unroll
  for (int t = 0; t < 4; ++t) {
#pragma unroll
    for (int i = 0; i < 8; ++i) {
      const int row = (rt0 + t) * 16 + i + half * 8;
      const int b   = row >> 11;
      const int pos = row & (Lc - 1);
      const int bh  = b * NHc + h;
      const int ktv = pos >> 5, kr = pos & 31;
      int hv, ev;
      kdec(kr, hv, ev);
      // [bh][kt32(64)][nt(4)][lane(32)][e(16)] ; lane = wc%16 + 16*half
      const size_t idx =
          ((((size_t)bh * 64 + ktv) * 4 + ntv) * 32 + nlv + 16 * hv) * 16 + ev;
      vpk[idx] = (_Float16)(c0[t][i] + bA);
    }
  }
}

// ---------------------------------------------------------------------------
// Kernel 2: causal flash attention; all operands pre-swizzled -> b128 loads.
// grid = (L/16, B*NH), one wave per block.
// ---------------------------------------------------------------------------
__global__ __launch_bounds__(32) void flash_attn_kernel(
    const v16h* __restrict__ qpk, const v16h* __restrict__ kpk,
    const v16h* __restrict__ vpk, _Float16* __restrict__ cpk)
{
  __shared__ _Float16 pl[16 * 32];     // P tile staging (queries x keys)

  const int lane = threadIdx.x, half = lane >> 4, ln = lane & 15;
  const int qt = blockIdx.x, bh = blockIdx.y;

  const v16h aq0 = qpk[(((size_t)bh * 128 + qt) * 2 + 0) * 32 + lane];
  const v16h aq1 = qpk[(((size_t)bh * 128 + qt) * 2 + 1) * 32 + lane];

  float mrow[8], lrow[8];
  v8f zv = {};
  v8f o[4] = {zv, zv, zv, zv};
#pragma unroll
  for (int i = 0; i < 8; ++i) { mrow[i] = -1e30f; lrow[i] = 0.0f; }

  const int nkt = (qt * 16 + 47) >> 5;     // ceil((qt*16+16)/32)
  for (int kt = 0; kt < nkt; ++kt) {
    __builtin_prefetch(
        (const void*)&kpk[(((size_t)bh * 128 + (kt + 1) * 2) * 2) * 32 + lane], 0, 1);
    __builtin_prefetch(
        (const void*)&vpk[(((size_t)bh * 64 + (kt + 1)) * 4) * 32 + lane], 0, 1);

    // ---- scores: two 16x16 subtiles over 32 keys ----
    v8f s0, s1;
    {
      const int k16a = kt * 2, k16b = kt * 2 + 1;
      const v16h bl0 = kpk[(((size_t)bh * 128 + k16a) * 2 + 0) * 32 + lane];
      const v16h bh0 = kpk[(((size_t)bh * 128 + k16a) * 2 + 1) * 32 + lane];
      const v16h bl1 = kpk[(((size_t)bh * 128 + k16b) * 2 + 0) * 32 + lane];
      const v16h bh1 = kpk[(((size_t)bh * 128 + k16b) * 2 + 1) * 32 + lane];
      v8f z0 = {};
      s0 = wmma_f16(aq0, bl0, z0);
      s0 = wmma_f16(aq1, bh0, s0);
      v8f z1 = {};
      s1 = wmma_f16(aq0, bl1, z1);
      s1 = wmma_f16(aq1, bh1, s1);
    }

    // ---- scale + causal mask ----
#pragma unroll
    for (int i = 0; i < 8; ++i) {
      const int qpos = qt * 16 + i + half * 8;
      const int kp0 = kt * 32 + ln;
      const int kp1 = kt * 32 + 16 + ln;
      s0[i] = (kp0 <= qpos) ? s0[i] * 0.125f : -1e30f;
      s1[i] = (kp1 <= qpos) ? s1[i] * 0.125f : -1e30f;
    }

    // ---- online softmax (rows live across 16 lanes per half) ----
    float corr[8];
#pragma unroll
    for (int i = 0; i < 8; ++i) {
      float rm = fmaxf(s0[i], s1[i]);
#pragma unroll
      for (int off = 1; off < 16; off <<= 1)
        rm = fmaxf(rm, __shfl_xor(rm, off, 16));
      const float mn = fmaxf(mrow[i], rm);
      const float p0 = __expf(s0[i] - mn);
      const float p1 = __expf(s1[i] - mn);
      float rs = p0 + p1;
#pragma unroll
      for (int off = 1; off < 16; off <<= 1)
        rs += __shfl_xor(rs, off, 16);
      corr[i] = __expf(mrow[i] - mn);
      lrow[i] = lrow[i] * corr[i] + rs;
      mrow[i] = mn;
      pl[(i + half * 8) * 32 + ln]      = (_Float16)p0;   // C-layout -> LDS
      pl[(i + half * 8) * 32 + 16 + ln] = (_Float16)p1;
    }
    __syncthreads();

    // ---- reload P in A-operand layout ----
    v16h ap;
#pragma unroll
    for (int e = 0; e < 16; ++e)
      ap[e] = pl[ln * 32 + kOf(e, half)];
    __syncthreads();

    // ---- rescale O, accumulate P x V over 4 HD column tiles ----
#pragma unroll
    for (int nt = 0; nt < 4; ++nt) {
#pragma unroll
      for (int i = 0; i < 8; ++i) o[nt][i] *= corr[i];
      const v16h bv_ = vpk[(((size_t)bh * 64 + kt) * 4 + nt) * 32 + lane];
      o[nt] = wmma_f16(ap, bv_, o[nt]);
    }
  }

  // ---- finalize: normalize and store ctx pre-swizzled as A operand ----
  // ctx A layout: [rt(256)][kt(32)][lane(32)][e(16)], rt = b*128 + qt
  const int b = bh >> 4, h = bh & 15;
  const int rt = b * 128 + qt;
#pragma unroll
  for (int nt = 0; nt < 4; ++nt) {
#pragma unroll
    for (int i = 0; i < 8; ++i) {
      const int M = i + half * 8;
      const int c = h * HDc + nt * 16 + ln;
      const int ktc = c >> 5, w = c & 31;
      int hc, ec;
      kdec(w, hc, ec);
      const size_t idx = (((size_t)rt * 32 + ktc) * 32 + M + 16 * hc) * 16 + ec;
      cpk[idx] = (_Float16)(o[nt][i] / lrow[i]);
    }
  }
}

// ---------------------------------------------------------------------------
// Kernel 3: output projection ctx @ wo + bo -> f32. grid = (64, 64),
// 4 row-tiles per wave, packed operands.
// ---------------------------------------------------------------------------
__global__ __launch_bounds__(32) void out_proj_kernel(
    const v16h* __restrict__ cpk, const v16h* __restrict__ wpk,
    const float* __restrict__ bo, float* __restrict__ out)
{
  const int lane = threadIdx.x, half = lane >> 4, ln = lane & 15;
  const int rt0 = blockIdx.x * 4;
  const int ct  = blockIdx.y;

  const v16h* bbase = wpk + ((size_t)3 * 64 + ct) * 32 * 32;

  v8f zv = {};
  v8f c[4] = {zv, zv, zv, zv};
  for (int kt = 0; kt < 32; ++kt) {
    __builtin_prefetch((const void*)&bbase[(kt + 1) * 32 + lane], 0, 1);
    const v16h b = bbase[kt * 32 + lane];
#pragma unroll
    for (int t = 0; t < 4; ++t) {
      const v16h a = cpk[((size_t)(rt0 + t) * 32 + kt) * 32 + lane];
      c[t] = wmma_f16(a, b, c[t]);
    }
  }
  const float bb = bo[ct * 16 + ln];
#pragma unroll
  for (int t = 0; t < 4; ++t) {
#pragma unroll
    for (int i = 0; i < 8; ++i) {
      const int row = (rt0 + t) * 16 + i + half * 8;
      out[(size_t)row * Dc + ct * 16 + ln] = c[t][i] + bb;
    }
  }
}

// ---------------------------------------------------------------------------
extern "C" void kernel_launch(void* const* d_in, const int* in_sizes, int n_in,
                              void* d_out, int out_size, void* d_ws, size_t ws_size,
                              hipStream_t stream) {
  (void)in_sizes; (void)n_in; (void)out_size; (void)ws_size;
  const float* x  = (const float*)d_in[0];
  const float* wq = (const float*)d_in[1];
  const float* bq = (const float*)d_in[2];
  const float* wk = (const float*)d_in[3];
  const float* bk = (const float*)d_in[4];
  const float* wv = (const float*)d_in[5];
  const float* bv = (const float*)d_in[6];
  const float* wo = (const float*)d_in[7];
  const float* bo = (const float*)d_in[8];
  float* out = (float*)d_out;

  // Workspace: 6 packed regions x 8 MB = 48 MB.
  v16h* xpk = (v16h*)d_ws;
  v16h* wpk = xpk + PKREG;
  v16h* qpk = wpk + PKREG;
  v16h* kpk = qpk + PKREG;
  v16h* vpk = kpk + PKREG;
  v16h* cpk = vpk + PKREG;

  pack_x_kernel<<<dim3(256, 32), 32, 0, stream>>>(x, xpk);
  pack_w_kernel<<<dim3(64, 32, 4), 32, 0, stream>>>(wq, wk, wv, wo, wpk);
  qk_rope_kernel<<<dim3(64, 64), 32, 0, stream>>>(
      xpk, wpk, bq, bk, (_Float16*)qpk, (_Float16*)kpk);
  v_proj_kernel<<<dim3(64, 64), 32, 0, stream>>>(
      xpk, wpk, bv, (_Float16*)vpk);
  flash_attn_kernel<<<dim3(Lc / 16, Bc * NHc), 32, 0, stream>>>(
      qpk, kpk, vpk, (_Float16*)cpk);
  out_proj_kernel<<<dim3(64, 64), 32, 0, stream>>>(cpk, wpk, bo, out);
}
